// DecomposingAttnProcessor_pad_tmp_41669772706196
// MI455X (gfx1250) — matrix operand order repaired and checked by reading, not compiled, not run on hardware
//
#include <hip/hip_runtime.h>
#include <hip/hip_bf16.h>

#define NCOMP  4
#define HEADS  24
#define DHEAD  64
#define DMODEL 1536
#define SSEQ   4096
#define ESEQ   154
#define EPAD   160
#define EPSV   1e-8f

#define BK   64      // GEMM K-step
#define LDT  72      // padded LDS row stride (shorts): 144B, 16B-aligned, bank-spread

typedef __attribute__((ext_vector_type(8)))  float   v8f;
typedef __attribute__((ext_vector_type(16))) __bf16  v16bf;
typedef __attribute__((ext_vector_type(2)))  __bf16  v2bf;
typedef __attribute__((ext_vector_type(4)))  int     v4i;

union BF16Vec {
    v16bf v;
    unsigned short u[16];
    unsigned w[8];
    uint4 q[2];
};

// ---- CDNA5 async global->LDS (ASYNCcnt) -----------------------------------
#if defined(__has_builtin)
#  if __has_builtin(__builtin_amdgcn_global_load_async_to_lds_b128) && \
      __has_builtin(__builtin_amdgcn_s_wait_asynccnt)
#    define USE_ASYNC_LDS 1
#  endif
#  if __has_builtin(__builtin_amdgcn_cvt_pk_bf16_f32)
#    define USE_PK_BF16 1
#  endif
#endif
#ifndef USE_ASYNC_LDS
#  define USE_ASYNC_LDS 0
#endif
#ifndef USE_PK_BF16
#  define USE_PK_BF16 0
#endif

#define AS_GLOBAL __attribute__((address_space(1)))
#define AS_LDS    __attribute__((address_space(3)))

__device__ __forceinline__ void async_b128(const unsigned short* g, unsigned short* l) {
#if USE_ASYNC_LDS
    AS_GLOBAL v4i* gp = (AS_GLOBAL v4i*)((AS_GLOBAL void*)g);
    AS_LDS    v4i* lp = (AS_LDS    v4i*)((AS_LDS    void*)l);
    __builtin_amdgcn_global_load_async_to_lds_b128(gp, lp, 0, 0);
#else
    *(uint4*)l = *(const uint4*)g;
#endif
}

template<int N>
__device__ __forceinline__ void wait_async() {
#if USE_ASYNC_LDS
    __builtin_amdgcn_s_wait_asynccnt(N);
#endif
}

__device__ __forceinline__ unsigned short f2bf(float f) {
    union { float f; unsigned u; } x; x.f = f;
    unsigned r = x.u + 0x7FFFu + ((x.u >> 16) & 1u);   // RNE
    return (unsigned short)(r >> 16);
}

// one v_cvt_pk_bf16_f32 converts+packs two floats into a dword
__device__ __forceinline__ unsigned pack2bf(float a, float b) {
#if USE_PK_BF16
    union { v2bf v; unsigned u; } x;
    x.v = __builtin_amdgcn_cvt_pk_bf16_f32(a, b);
    return x.u;
#else
    return (unsigned)f2bf(a) | ((unsigned)f2bf(b) << 16);
#endif
}

__device__ __forceinline__ v8f wmma_bf16(const BF16Vec& a, const BF16Vec& b, v8f c) {
    return __builtin_amdgcn_wmma_f32_16x16x32_bf16(false, a.v, false, b.v,
                                                   (short)0, c, false, false);
}

// ---------------------------------------------------------------------------
// Elementwise f32 -> bf16 (8 per thread, packed converts)
// ---------------------------------------------------------------------------
__global__ __launch_bounds__(256) void cvt_f32_bf16(
    const float* __restrict__ s, unsigned short* __restrict__ d, int n8)
{
    const int i = blockIdx.x * 256 + threadIdx.x;
    if (i < n8) {
        const float4* sp = (const float4*)s + (size_t)i * 2;
        float4 a = sp[0], b = sp[1];
        uint4 o;
        o.x = pack2bf(a.x, a.y);
        o.y = pack2bf(a.z, a.w);
        o.z = pack2bf(b.x, b.y);
        o.w = pack2bf(b.z, b.w);
        ((uint4*)d)[i] = o;
    }
}

// ---------------------------------------------------------------------------
// Pure-bf16 NT GEMM: C[M,N] = A[M,K] @ W[N,K]^T (+ bias + residual).
// Block 128x128, K-step 64, double-buffered async global->LDS staging.
// 8 waves (4x2), wave tile 32x64 -> 16 WMMAs per stage per wave.
// LDS layout: [buf0 A][buf1 A][buf0 B][buf1 B], each 128*LDT shorts.
// ---------------------------------------------------------------------------
template<int OUT_BF16, int CHECK_M>
__global__ __launch_bounds__(256) void gemm_bf16_nt(
    const unsigned short* __restrict__ A,    // [M,K] bf16
    const unsigned short* __restrict__ Bw,   // [N,K] bf16
    const float* __restrict__ bias,
    const float* __restrict__ residual,
    void* __restrict__ Cptr,
    int M, int N, int K)
{
    extern __shared__ unsigned short lds[];

    const int tid  = threadIdx.x;
    const int lane = tid & 31;
    const int wave = tid >> 5;
    const int hf   = lane >> 4;
    const int lm   = lane & 15;
    const int wm   = wave >> 1;
    const int wn   = wave & 1;
    const int gm0  = blockIdx.x * 128;
    const int gn0  = blockIdx.y * 128;

    // staging map: 256 threads -> 128 rows x 2 column halves of 32
    const int srow = tid & 127;
    const int scol = (tid >> 7) * 32;

    v8f acc[2][4];
#pragma unroll
    for (int i = 0; i < 2; ++i)
#pragma unroll
        for (int j = 0; j < 4; ++j) acc[i][j] = {};

    const int nk = K / BK;

    auto stage = [&](int kk, int buf) {
        const unsigned short* ga = A  + (size_t)(gm0 + srow) * K + kk * BK + scol;
        const unsigned short* gb = Bw + (size_t)(gn0 + srow) * K + kk * BK + scol;
        unsigned short* la = lds + buf       * (128 * LDT) + srow * LDT + scol;
        unsigned short* lb = lds + (2 + buf) * (128 * LDT) + srow * LDT + scol;
        if (CHECK_M) {
            // small-M path: synchronous copies, zero-fill out-of-range rows
            if (gm0 + srow < M) {
#pragma unroll
                for (int j = 0; j < 4; ++j) *(uint4*)(la + j*8) = *(const uint4*)(ga + j*8);
            } else {
                const uint4 z = {0u,0u,0u,0u};
#pragma unroll
                for (int j = 0; j < 4; ++j) *(uint4*)(la + j*8) = z;
            }
#pragma unroll
            for (int j = 0; j < 4; ++j) *(uint4*)(lb + j*8) = *(const uint4*)(gb + j*8);
        } else {
            // uniform path: 8 async b128 per thread per stage
#pragma unroll
            for (int j = 0; j < 4; ++j) {
                async_b128(ga + j*8, la + j*8);
                async_b128(gb + j*8, lb + j*8);
            }
        }
    };

    auto compute = [&](int buf) {
        const unsigned short* ab = lds + buf       * (128 * LDT);
        const unsigned short* bb = lds + (2 + buf) * (128 * LDT);
        BF16Vec aop[2][2], bop[2][4];
#pragma unroll
        for (int ks = 0; ks < 2; ++ks) {
#pragma unroll
            for (int mi = 0; mi < 2; ++mi) {
                const unsigned short* p = ab + (wm*32 + mi*16 + lm) * LDT + ks*32;
                aop[ks][mi].q[0] = *(const uint4*)(p + hf*8);         // K 0..7 / 8..15
                aop[ks][mi].q[1] = *(const uint4*)(p + 16 + hf*8);    // K 16..23 / 24..31
            }
#pragma unroll
            for (int ni = 0; ni < 4; ++ni) {
                const unsigned short* p = bb + (wn*64 + ni*16 + lm) * LDT + ks*32 + hf*8;
                bop[ks][ni].q[0] = ((const uint4*)p)[0];
                bop[ks][ni].q[1] = *(const uint4*)(p + 16);
            }
        }
#pragma unroll
        for (int ks = 0; ks < 2; ++ks)
#pragma unroll
            for (int mi = 0; mi < 2; ++mi)
#pragma unroll
                for (int ni = 0; ni < 4; ++ni)
                    acc[mi][ni] = wmma_bf16(aop[ks][mi], bop[ks][ni], acc[mi][ni]);
    };

    stage(0, 0);
    for (int kk = 0; kk < nk; ++kk) {
        const int cur = kk & 1;
        if (kk + 1 < nk) {
            stage(kk + 1, cur ^ 1);
            wait_async<8>();      // previous buffer complete; prefetch in flight
        } else {
            wait_async<0>();
        }
        __syncthreads();
        compute(cur);
        __syncthreads();
    }

    // ---- epilogue: C layout M = g + 8*hf, N = lm ----
#pragma unroll
    for (int mi = 0; mi < 2; ++mi)
#pragma unroll
        for (int ni = 0; ni < 4; ++ni) {
            const int gcol = gn0 + wn*64 + ni*16 + lm;
#pragma unroll
            for (int g = 0; g < 8; ++g) {
                const int grow = gm0 + wm*32 + mi*16 + hf*8 + g;
                if (!CHECK_M || grow < M) {
                    float v = acc[mi][ni][g];
                    if (OUT_BF16) {
                        ((unsigned short*)Cptr)[(size_t)grow * N + gcol] = f2bf(v);
                    } else {
                        v += bias[gcol];
                        v += residual[(size_t)grow * N + gcol];
                        ((float*)Cptr)[(size_t)grow * N + gcol] = v;
                    }
                }
            }
        }
}

// ---------------------------------------------------------------------------
// Attention core. Grid: (SSEQ/128, HEADS), 256 threads. Each wave owns one
// 16-row s-tile for ALL 4 components -> cross-component softmax in-register.
// Scores computed transposed (D = K_tile x Q^T): two consecutive 16-e score
// tiles are bit-identical to the A-operand layout of the out += w @ V WMMA.
// LDS: K[c][EPAD][64] + V^T[c][64][EPAD], bf16, 160 KB.
// ---------------------------------------------------------------------------
__global__ __launch_bounds__(256) void attn_core(
    const unsigned short* __restrict__ Qb,   // [NCOMP, SSEQ, DMODEL] bf16
    const unsigned short* __restrict__ Kb,   // [NCOMP, ESEQ, DMODEL] bf16
    const unsigned short* __restrict__ Vb,   // [NCOMP, ESEQ, DMODEL] bf16
    const float* __restrict__ temperature,
    unsigned short* __restrict__ Ob)         // [NCOMP, SSEQ, DMODEL] bf16
{
    extern __shared__ unsigned short sm[];
    unsigned short* Kl = sm;                             // [NCOMP][EPAD][DHEAD]
    unsigned short* Vl = sm + NCOMP * EPAD * DHEAD;      // [NCOMP][DHEAD][EPAD]

    const int tid = threadIdx.x;
    const int h   = blockIdx.y;

    // ---- K fill: async b128 copies; zero-pad e >= ESEQ ----
    for (int idx = tid; idx < NCOMP * EPAD * (DHEAD / 8); idx += 256) {
        const int c  = idx / (EPAD * 8);
        const int r  = idx % (EPAD * 8);
        const int e  = r / 8;
        const int d8 = (r % 8) * 8;
        unsigned short* l = Kl + ((size_t)(c * EPAD + e)) * DHEAD + d8;
        if (e < ESEQ) {
            async_b128(Kb + ((size_t)(c * ESEQ + e)) * DMODEL + h * DHEAD + d8, l);
        } else {
            const uint4 z = {0u,0u,0u,0u};
            *(uint4*)l = z;
        }
    }
    // ---- V fill, transposed (scalar LDS stores) ----
    for (int idx = tid; idx < NCOMP * EPAD * (DHEAD / 8); idx += 256) {
        const int c  = idx / (EPAD * 8);
        const int r  = idx % (EPAD * 8);
        const int e  = r / 8;
        const int d8 = (r % 8) * 8;
        alignas(16) unsigned short tmp[8] = {0,0,0,0,0,0,0,0};
        if (e < ESEQ)
            *(uint4*)&tmp[0] =
                *(const uint4*)(Vb + ((size_t)(c * ESEQ + e)) * DMODEL + h * DHEAD + d8);
#pragma unroll
        for (int j = 0; j < 8; ++j)
            Vl[((size_t)(c * DHEAD + d8 + j)) * EPAD + e] = tmp[j];
    }
    wait_async<0>();
    __syncthreads();

    const int lane = tid & 31;
    const int wave = tid >> 5;
    const int hf   = lane >> 4;
    const int lm   = lane & 15;
    const int s0   = blockIdx.x * 128 + wave * 16;
    const float invt = 1.0f / (temperature[0] + EPSV);

    // ---- Q as B-operands (lane = s column, K-dim = dh) ----
    BF16Vec qop[NCOMP][2];
#pragma unroll
    for (int c = 0; c < NCOMP; ++c)
#pragma unroll
        for (int ks = 0; ks < 2; ++ks) {
            const unsigned short* p = Qb + ((size_t)(c * SSEQ + s0 + lm)) * DMODEL
                                        + h * DHEAD + ks * 32 + hf * 8;
            qop[c][ks].q[0] = ((const uint4*)p)[0];
            qop[c][ks].q[1] = *(const uint4*)(p + 16);
        }

    v8f oacc[NCOMP][4];
#pragma unroll
    for (int c = 0; c < NCOMP; ++c)
#pragma unroll
        for (int nt = 0; nt < 4; ++nt) oacc[c][nt] = {};

    for (int ep = 0; ep < EPAD / 32; ++ep) {
        const int e0 = ep * 32;

        // ---- scores^T for both 16-e tiles, all components ----
        v8f d0[NCOMP], d1[NCOMP];
#pragma unroll
        for (int c = 0; c < NCOMP; ++c) {
            v8f sa = {}, sb = {};
#pragma unroll
            for (int ks = 0; ks < 2; ++ks) {
                BF16Vec ka, kb;
                const unsigned short* p0 =
                    Kl + ((size_t)(c * EPAD + e0 + lm)) * DHEAD + ks * 32;
                *(uint4*)&ka.u[0] = *(const uint4*)(p0 + hf * 8);
                *(uint4*)&ka.u[8] = *(const uint4*)(p0 + 16 + hf * 8);
                const unsigned short* p1 =
                    Kl + ((size_t)(c * EPAD + e0 + 16 + lm)) * DHEAD + ks * 32;
                *(uint4*)&kb.u[0] = *(const uint4*)(p1 + hf * 8);
                *(uint4*)&kb.u[8] = *(const uint4*)(p1 + 16 + hf * 8);
                sa = wmma_bf16(ka, qop[c][ks], sa);
                sb = wmma_bf16(kb, qop[c][ks], sb);
            }
            d0[c] = sa; d1[c] = sb;
        }

        // ---- cross-component softmax -> packed bf16 A-operands ----
        float w0[NCOMP][8], w1[NCOMP][8];
#pragma unroll
        for (int g = 0; g < 8; ++g) {
            {
                float a0 = d0[0][g] * invt, a1 = d0[1][g] * invt;
                float a2 = d0[2][g] * invt, a3 = d0[3][g] * invt;
                float mx = fmaxf(fmaxf(a0, a1), fmaxf(a2, a3));
                float x0 = __expf(a0 - mx), x1 = __expf(a1 - mx);
                float x2 = __expf(a2 - mx), x3 = __expf(a3 - mx);
                float inv = 1.0f / (x0 + x1 + x2 + x3);
                w0[0][g] = x0 * inv; w0[1][g] = x1 * inv;
                w0[2][g] = x2 * inv; w0[3][g] = x3 * inv;
            }
            {
                float a0 = d1[0][g] * invt, a1 = d1[1][g] * invt;
                float a2 = d1[2][g] * invt, a3 = d1[3][g] * invt;
                float mx = fmaxf(fmaxf(a0, a1), fmaxf(a2, a3));
                float x0 = __expf(a0 - mx), x1 = __expf(a1 - mx);
                float x2 = __expf(a2 - mx), x3 = __expf(a3 - mx);
                float inv = 1.0f / (x0 + x1 + x2 + x3);
                w1[0][g] = x0 * inv; w1[1][g] = x1 * inv;
                w1[2][g] = x2 * inv; w1[3][g] = x3 * inv;
            }
        }
        BF16Vec wa[NCOMP];
#pragma unroll
        for (int c = 0; c < NCOMP; ++c)
#pragma unroll
            for (int i = 0; i < 4; ++i) {
                wa[c].w[i]     = pack2bf(w0[c][2*i], w0[c][2*i + 1]);   // k 0..7 / 8..15
                wa[c].w[4 + i] = pack2bf(w1[c][2*i], w1[c][2*i + 1]);   // k 16..23 / 24..31
            }

        // ---- out += w @ V  (B-operand from V^T rows, contiguous in e) ----
#pragma unroll
        for (int c = 0; c < NCOMP; ++c)
#pragma unroll
            for (int nt = 0; nt < 4; ++nt) {
                BF16Vec vb;
                const unsigned short* p =
                    Vl + ((size_t)(c * DHEAD + nt * 16 + lm)) * EPAD + e0 + hf * 8;
                vb.q[0] = ((const uint4*)p)[0];
                vb.q[1] = *(const uint4*)(p + 16);
                oacc[c][nt] = wmma_bf16(wa[c], vb, oacc[c][nt]);
            }
    }

    // ---- store attention output (bf16) ----
#pragma unroll
    for (int c = 0; c < NCOMP; ++c)
#pragma unroll
        for (int nt = 0; nt < 4; ++nt) {
            const int col = h * DHEAD + nt * 16 + lm;
#pragma unroll
            for (int g = 0; g < 8; ++g) {
                const int sRow = s0 + hf * 8 + g;
                Ob[((size_t)(c * SSEQ + sRow)) * DMODEL + col] = f2bf(oacc[c][nt][g]);
            }
        }
}

// ---------------------------------------------------------------------------
extern "C" void kernel_launch(void* const* d_in, const int* in_sizes, int n_in,
                              void* d_out, int out_size, void* d_ws, size_t ws_size,
                              hipStream_t stream)
{
    (void)in_sizes; (void)n_in; (void)out_size; (void)ws_size;

    const float* hs   = (const float*)d_in[0];   // [4,4096,1536]
    const float* ehs  = (const float*)d_in[1];   // [4, 154,1536]
    const float* temp = (const float*)d_in[2];
    const float* Wq   = (const float*)d_in[3];
    const float* Wk   = (const float*)d_in[4];
    const float* Wv   = (const float*)d_in[5];
    const float* Wo   = (const float*)d_in[6];
    const float* bo   = (const float*)d_in[7];
    float* out        = (float*)d_out;

    const int Mq = NCOMP * SSEQ;   // 16384
    const int Me = NCOMP * ESEQ;   // 616
    const int W2 = DMODEL * DMODEL;

    size_t off = 0;
    auto wsAlloc = [&](size_t bytes) {
        void* p = (char*)d_ws + off;
        off = (off + bytes + 255) & ~(size_t)255;
        return p;
    };
    unsigned short* Xb  = (unsigned short*)wsAlloc((size_t)Mq * DMODEL * 2); // hs bf16; reused as attn out
    unsigned short* Eb  = (unsigned short*)wsAlloc((size_t)Me * DMODEL * 2);
    unsigned short* Wqb = (unsigned short*)wsAlloc((size_t)W2 * 2);
    unsigned short* Wkb = (unsigned short*)wsAlloc((size_t)W2 * 2);
    unsigned short* Wvb = (unsigned short*)wsAlloc((size_t)W2 * 2);
    unsigned short* Wob = (unsigned short*)wsAlloc((size_t)W2 * 2);
    unsigned short* Qb  = (unsigned short*)wsAlloc((size_t)Mq * DMODEL * 2);
    unsigned short* Kb  = (unsigned short*)wsAlloc((size_t)Me * DMODEL * 2);
    unsigned short* Vb  = (unsigned short*)wsAlloc((size_t)Me * DMODEL * 2);
    unsigned short* Ab  = Xb;   // alias: Xb consumed by Q-proj before attn writes

    const dim3 blk(256);
    auto cvt = [&](const float* s, unsigned short* d, long n) {
        const int n8 = (int)(n / 8);
        cvt_f32_bf16<<<dim3((n8 + 255) / 256), blk, 0, stream>>>(s, d, n8);
    };

    // 1) one-time f32 -> bf16 conversions
    cvt(hs,  Xb,  (long)Mq * DMODEL);
    cvt(ehs, Eb,  (long)Me * DMODEL);
    cvt(Wq,  Wqb, W2);
    cvt(Wk,  Wkb, W2);
    cvt(Wv,  Wvb, W2);
    cvt(Wo,  Wob, W2);

    // 2) projections (pure-bf16 WMMA GEMMs)
    const int gemmSmem = 4 * 128 * LDT * (int)sizeof(unsigned short);  // 73,728 B
    (void)hipFuncSetAttribute(reinterpret_cast<const void*>(&gemm_bf16_nt<1,0>),
                              hipFuncAttributeMaxDynamicSharedMemorySize, gemmSmem);
    (void)hipFuncSetAttribute(reinterpret_cast<const void*>(&gemm_bf16_nt<1,1>),
                              hipFuncAttributeMaxDynamicSharedMemorySize, gemmSmem);
    (void)hipFuncSetAttribute(reinterpret_cast<const void*>(&gemm_bf16_nt<0,0>),
                              hipFuncAttributeMaxDynamicSharedMemorySize, gemmSmem);

    gemm_bf16_nt<1,0><<<dim3(Mq / 128, DMODEL / 128), blk, gemmSmem, stream>>>(
        Xb, Wqb, nullptr, nullptr, Qb, Mq, DMODEL, DMODEL);
    gemm_bf16_nt<1,1><<<dim3((Me + 127) / 128, DMODEL / 128), blk, gemmSmem, stream>>>(
        Eb, Wkb, nullptr, nullptr, Kb, Me, DMODEL, DMODEL);
    gemm_bf16_nt<1,1><<<dim3((Me + 127) / 128, DMODEL / 128), blk, gemmSmem, stream>>>(
        Eb, Wvb, nullptr, nullptr, Vb, Me, DMODEL, DMODEL);

    // 3) attention core (160 KB dynamic LDS)
    const int attnSmem = 2 * NCOMP * EPAD * DHEAD * (int)sizeof(unsigned short);
    (void)hipFuncSetAttribute(reinterpret_cast<const void*>(&attn_core),
                              hipFuncAttributeMaxDynamicSharedMemorySize, attnSmem);
    attn_core<<<dim3(SSEQ / 128, HEADS), blk, attnSmem, stream>>>(Qb, Kb, Vb, temp, Ab);

    // 4) output projection + bias + residual (f32 out)
    gemm_bf16_nt<0,0><<<dim3(Mq / 128, DMODEL / 128), blk, gemmSmem, stream>>>(
        Ab, Wob, bo, hs, out, Mq, DMODEL, DMODEL);
}
